// TitansMemory_6055903888125
// MI455X (gfx1250) — matrix operand order repaired and checked by reading, not compiled, not run on hardware
//
#include <hip/hip_runtime.h>

#define BATCH 8192
#define CDIM  512
#define SLOTS 32
#define TB    16          // batch rows per block
#define NT32  32          // CDIM/16 n-tiles

typedef __attribute__((ext_vector_type(16))) _Float16 v16h;
typedef __attribute__((ext_vector_type(8)))  _Float16 v8h;
typedef __attribute__((ext_vector_type(8)))  float    v8f;
typedef __attribute__((ext_vector_type(4)))  float    v4f;

// ---------------------------------------------------------------------------
// Pack fp32 weight W (N rows x K cols, row-major, used as B = W^T in K x N)
// into f16 WMMA-B tile order: tile (ks,ns) holds K rows ks*32..+31, N cols
// ns*16..+15; lane (hi*16 + n) owns N col n, 16 K values in table order (one 32B load).
// ---------------------------------------------------------------------------
__global__ void pack_w_f16(const float* __restrict__ W, _Float16* __restrict__ out,
                           int N, int K) {
  int idx = blockIdx.x * 256 + threadIdx.x;
  if (idx >= N * K) return;
  int n = idx / K, k = idx - n * K;
  int ks = k >> 5, kl = k & 31, ns = n >> 4, nl = n & 15;
  int NTn = N >> 4;
  int bl = (((kl >> 3) & 1) << 4) + nl, bs = (kl & 7) + ((kl >> 4) << 3);
  out[(((size_t)(ks * NTn + ns) * 32 + bl) << 4) + bs] = (_Float16)W[idx];
}

// A-fragment: 16-bit A 16x32 layout -> lane (m = lane&15, hi = lane>>4)
// holds K = s*32 + hi*8 + {0..7} and + {16..23}: two contiguous b128 LDS reads.
__device__ __forceinline__ v16h load_a_frag(const _Float16* row, int hi) {
  const _Float16* p = row + hi * 8;
  v8h lo = *(const v8h*)p;
  v8h hh = *(const v8h*)(p + 16);
  v16h a;
#pragma unroll
  for (int i = 0; i < 8; ++i) { a[i] = lo[i]; a[i + 8] = hh[i]; }
  return a;
}

// 16 x CDIM GEMM, 8 waves x 4 persistent n-tiles, software-pipelined B loads.
template <int KS>
__device__ __forceinline__ void gemm16(const _Float16* sA0, const _Float16* sA1,
                                       const _Float16* __restrict__ Bpk,
                                       int wave, int lane, v8f acc[4]) {
  const int m0 = lane & 15;
  const int hi = lane >> 4;
  const _Float16* bbase = Bpk + ((size_t)lane << 4);
  v16h bcur[4], bnxt[4];
#pragma unroll
  for (int j = 0; j < 4; ++j)
    bcur[j] = *(const v16h*)(bbase + ((size_t)(wave + 8 * j) << 9));
#pragma unroll
  for (int s = 0; s < KS; ++s) {
    const _Float16* arow = (s < 16) ? (sA0 + m0 * CDIM + s * 32)
                                    : (sA1 + m0 * CDIM + (s - 16) * 32);
    v16h a = load_a_frag(arow, hi);
    if (s + 1 < KS) {
#pragma unroll
      for (int j = 0; j < 4; ++j)
        bnxt[j] = *(const v16h*)(bbase + ((size_t)((s + 1) * NT32 + wave + 8 * j) << 9));
    }
#pragma unroll
    for (int j = 0; j < 4; ++j)
      acc[j] = __builtin_amdgcn_wmma_f32_16x16x32_f16(false, a, false, bcur[j],
                                                      (short)0, acc[j], false, false);
#pragma unroll
    for (int j = 0; j < 4; ++j) bcur[j] = bnxt[j];
  }
}

__launch_bounds__(256)
__global__ void titans_fused(const float* __restrict__ h, const float* __restrict__ pooled,
                             const float* __restrict__ Mm, const float* __restrict__ Mk,
                             const float* __restrict__ bgate, const float* __restrict__ eta,
                             const _Float16* __restrict__ Wq, const _Float16* __restrict__ Wk,
                             const _Float16* __restrict__ Wv, const _Float16* __restrict__ Wg,
                             float* __restrict__ out0, float* __restrict__ outM,
                             float* __restrict__ outK) {
  __shared__ __align__(16) _Float16 sXa[TB * CDIM];   // h f16, later retrieved f16
  __shared__ __align__(16) _Float16 sXb[TB * CDIM];   // pooled f16
  __shared__ __align__(16) _Float16 sQ16[TB * CDIM];  // query f16
  __shared__ __align__(16) _Float16 sK16[TB * CDIM];  // write_k f16
  __shared__ __align__(16) float sV[TB * CDIM];       // write_v, later u
  __shared__ __align__(16) float sR[TB * CDIM];       // retrieved f32
  __shared__ float sAttn[TB * SLOTS];
  __shared__ float sWw[TB * SLOTS];
  __shared__ float sEta[CDIM];
  __shared__ float sBg[CDIM];
  // total LDS: 4*16KB + 2*32KB + ~8KB = ~136KB -> 2 workgroups / WGP

  const int t = threadIdx.x;
  const int wave = t >> 5;
  const int lane = t & 31;
  const int m0 = lane & 15;
  const int hi = lane >> 4;
  const int b0 = blockIdx.x * TB;
  const float inv_sqrt_c = 0.044194173824159216f;   // 1/sqrt(512)

  // ---- phase 0: stage inputs -------------------------------------------
  for (int i = t; i < CDIM; i += 256) {
    sEta[i] = 1.0f / (1.0f + __expf(-eta[i]));
    sBg[i] = bgate[i];
  }
  for (int i = t; i < TB * CDIM; i += 256) {
    int m = i >> 9, c = i & (CDIM - 1);
    size_t g = (size_t)(b0 + m) * CDIM + c;
    sXa[i] = (_Float16)h[g];
    sXb[i] = (_Float16)pooled[g];
  }
  __syncthreads();

  // ---- phase 1: q, write_k, write_v GEMMs (WMMA, pipelined) -------------
  {
    v8f acc[4] = {};
    gemm16<16>(sXa, sXa, Wq, wave, lane, acc);
#pragma unroll
    for (int j = 0; j < 4; ++j)
#pragma unroll
      for (int r = 0; r < 8; ++r)
        sQ16[(r + 8 * hi) * CDIM + (wave + 8 * j) * 16 + m0] = (_Float16)acc[j][r];
  }
  {
    v8f acc[4] = {};
    gemm16<16>(sXb, sXb, Wk, wave, lane, acc);
#pragma unroll
    for (int j = 0; j < 4; ++j)
#pragma unroll
      for (int r = 0; r < 8; ++r)
        sK16[(r + 8 * hi) * CDIM + (wave + 8 * j) * 16 + m0] = (_Float16)acc[j][r];
  }
  {
    v8f acc[4] = {};
    gemm16<16>(sXb, sXb, Wv, wave, lane, acc);
#pragma unroll
    for (int j = 0; j < 4; ++j)
#pragma unroll
      for (int r = 0; r < 8; ++r)
        sV[(r + 8 * hi) * CDIM + (wave + 8 * j) * 16 + m0] = acc[j][r];
  }
  __syncthreads();

  // ---- phase 2: logits (stream M_keys once; lane owns c = lane*16+j) ----
  for (int rr = 0; rr < 2; ++rr) {
    int m = wave * 2 + rr;
    float qv[16], wv[16];
    {
      const v8h* qh = (const v8h*)&sQ16[m * CDIM + lane * 16];
      const v8h* kh = (const v8h*)&sK16[m * CDIM + lane * 16];
      v8h q0 = qh[0], q1 = qh[1], k0 = kh[0], k1 = kh[1];
#pragma unroll
      for (int j = 0; j < 8; ++j) {
        qv[j] = (float)q0[j]; qv[8 + j] = (float)q1[j];
        wv[j] = (float)k0[j]; wv[8 + j] = (float)k1[j];
      }
    }
    size_t base = (size_t)(b0 + m) * SLOTS * CDIM;
#pragma unroll 2
    for (int s = 0; s < SLOTS; ++s) {
      const v4f* mk4 = (const v4f*)(Mk + base + (size_t)s * CDIM) + lane * 4;
      float a0 = 0.0f, a1 = 0.0f;
#pragma unroll
      for (int i = 0; i < 4; ++i) {
        v4f kv = mk4[i];
#pragma unroll
        for (int c = 0; c < 4; ++c) {
          a0 += kv[c] * qv[4 * i + c];
          a1 += kv[c] * wv[4 * i + c];
        }
      }
#pragma unroll
      for (int off = 16; off > 0; off >>= 1) {
        a0 += __shfl_xor(a0, off, 32);
        a1 += __shfl_xor(a1, off, 32);
      }
      if (lane == 0) { sAttn[m * SLOTS + s] = a0; sWw[m * SLOTS + s] = a1; }
    }
  }
  __syncthreads();

  // ---- phase 3: dual softmax over 32 slots ------------------------------
  for (int j = wave; j < 32; j += 8) {
    float* L = (j < 16) ? &sAttn[j * SLOTS] : &sWw[(j - 16) * SLOTS];
    float x = L[lane] * inv_sqrt_c;
    float mx = x;
#pragma unroll
    for (int off = 16; off > 0; off >>= 1) mx = fmaxf(mx, __shfl_xor(mx, off, 32));
    float e = __expf(x - mx);
    float ssum = e;
#pragma unroll
    for (int off = 16; off > 0; off >>= 1) ssum += __shfl_xor(ssum, off, 32);
    L[lane] = e / ssum;
  }
  __syncthreads();

  // ---- phase 4: retrieved = attn . M (stream M once) --------------------
  for (int rr = 0; rr < 2; ++rr) {
    int m = wave * 2 + rr;
    size_t base = (size_t)(b0 + m) * SLOTS * CDIM;
    v4f acc4[4] = {};
#pragma unroll 2
    for (int s = 0; s < SLOTS; ++s) {
      float w = sAttn[m * SLOTS + s];
      const v4f* Mp4 = (const v4f*)(Mm + base + (size_t)s * CDIM) + lane * 4;
#pragma unroll
      for (int i = 0; i < 4; ++i) {
        v4f mv = Mp4[i];
#pragma unroll
        for (int c = 0; c < 4; ++c) acc4[i][c] += w * mv[c];
      }
    }
    v8h h0, h1;
#pragma unroll
    for (int i = 0; i < 4; ++i) {
      *(v4f*)&sR[m * CDIM + lane * 16 + i * 4] = acc4[i];
#pragma unroll
      for (int c = 0; c < 4; ++c) {
        _Float16 hv = (_Float16)acc4[i][c];
        if (i < 2) h0[i * 4 + c] = hv; else h1[(i - 2) * 4 + c] = hv;
      }
    }
    *(v8h*)&sXa[m * CDIM + lane * 16] = h0;       // gate A, low-K half
    *(v8h*)&sXa[m * CDIM + lane * 16 + 8] = h1;
  }
  __syncthreads();

  // ---- phase 5: gate GEMM (K=1024, A = [retrieved | pooled]) + epilogue -
  {
    v8f acc[4] = {};
    gemm16<32>(sXa, sXb, Wg, wave, lane, acc);
#pragma unroll
    for (int j = 0; j < 4; ++j) {
#pragma unroll
      for (int r = 0; r < 8; ++r) {
        int m = r + 8 * hi, n = (wave + 8 * j) * 16 + m0;
        float sp = 1.0f / (1.0f + __expf(-(acc[j][r] + sBg[n])));          // surprise
        float u = sEta[n] * sp * (sV[m * CDIM + n] - sR[m * CDIM + n]);    // update vec
        sV[m * CDIM + n] = u;
      }
    }
  }
  __syncthreads();

  // ---- phase 6: outputs; M/M_keys re-reads are last-use (NT), stores NT --
  for (int i4 = t; i4 < TB * CDIM / 4; i4 += 256) {
    int m = i4 >> 7, c4 = i4 & 127;
    size_t g4 = (size_t)(b0 + m) * (CDIM / 4) + c4;
    v4f hv = ((const v4f*)h)[g4];
    v4f rv = *(const v4f*)&sR[m * CDIM + c4 * 4];
    v4f o = hv + rv;
    __builtin_nontemporal_store(o, &((v4f*)out0)[g4]);
  }
  for (int rr = 0; rr < 2; ++rr) {
    int m = wave * 2 + rr;
    size_t base = (size_t)(b0 + m) * SLOTS * CDIM;
    for (int s = 0; s < SLOTS; ++s) {
      float ww = sWw[m * SLOTS + s];
      float wlr = ww * 0.01f;                                  // KEY_LR
      const v4f* Mp = (const v4f*)(Mm + base + (size_t)s * CDIM);
      const v4f* Kp = (const v4f*)(Mk + base + (size_t)s * CDIM);
      v4f* Mo = (v4f*)(outM + base + (size_t)s * CDIM);
      v4f* Ko = (v4f*)(outK + base + (size_t)s * CDIM);
#pragma unroll
      for (int j = 0; j < 4; ++j) {
        int j4 = lane + 32 * j;
        v4f mv = __builtin_nontemporal_load(&Mp[j4]);
        v4f kv = __builtin_nontemporal_load(&Kp[j4]);
        v4f uv = *(const v4f*)&sV[m * CDIM + j4 * 4];
        v4f wk;
        const _Float16* kp16 = &sK16[m * CDIM + j4 * 4];
#pragma unroll
        for (int c = 0; c < 4; ++c) wk[c] = (float)kp16[c];
        mv += ww * uv;
        kv += wlr * (wk - kv);
        __builtin_nontemporal_store(mv, &Mo[j4]);
        __builtin_nontemporal_store(kv, &Ko[j4]);
      }
    }
  }
}

extern "C" void kernel_launch(void* const* d_in, const int* in_sizes, int n_in,
                              void* d_out, int out_size, void* d_ws, size_t ws_size,
                              hipStream_t stream) {
  (void)in_sizes; (void)n_in; (void)out_size; (void)ws_size;
  const float* h      = (const float*)d_in[0];
  const float* pooled = (const float*)d_in[1];
  const float* Mm     = (const float*)d_in[2];
  const float* Mk     = (const float*)d_in[3];
  const float* Wq     = (const float*)d_in[4];
  const float* Wk     = (const float*)d_in[5];
  const float* Wv     = (const float*)d_in[6];
  const float* Wg     = (const float*)d_in[7];
  const float* bg     = (const float*)d_in[8];
  const float* eta    = (const float*)d_in[9];

  _Float16* wq16 = (_Float16*)d_ws;
  _Float16* wk16 = wq16 + (size_t)CDIM * CDIM;
  _Float16* wv16 = wk16 + (size_t)CDIM * CDIM;
  _Float16* wg16 = wv16 + (size_t)CDIM * CDIM;

  const int nqkv = CDIM * CDIM;
  const int ngate = CDIM * 2 * CDIM;
  pack_w_f16<<<(nqkv + 255) / 256, 256, 0, stream>>>(Wq, wq16, CDIM, CDIM);
  pack_w_f16<<<(nqkv + 255) / 256, 256, 0, stream>>>(Wk, wk16, CDIM, CDIM);
  pack_w_f16<<<(nqkv + 255) / 256, 256, 0, stream>>>(Wv, wv16, CDIM, CDIM);
  pack_w_f16<<<(ngate + 255) / 256, 256, 0, stream>>>(Wg, wg16, CDIM, 2 * CDIM);

  float* out0 = (float*)d_out;
  float* outM = out0 + (size_t)BATCH * CDIM;
  float* outK = outM + (size_t)BATCH * SLOTS * CDIM;

  titans_fused<<<BATCH / TB, 256, 0, stream>>>(h, pooled, Mm, Mk, bg, eta,
                                               wq16, wk16, wv16, wg16,
                                               out0, outM, outK);
}
